// CovaMResnet56_22926535426720
// MI455X (gfx1250) — matrix operand order repaired and checked by reading, not compiled, not run on hardware
//
#include <hip/hip_runtime.h>

// ---------------------------------------------------------------------------
// CDNA5 (gfx1250) types for WMMA bf16 + TDM descriptors
// ---------------------------------------------------------------------------
typedef __bf16 bf16_t;
typedef __attribute__((ext_vector_type(8)))  __bf16 v8bf;
typedef __attribute__((ext_vector_type(16))) __bf16 v16bf;
typedef __attribute__((ext_vector_type(8)))  float  v8f;
typedef __attribute__((ext_vector_type(4)))  unsigned int u32x4;
typedef __attribute__((ext_vector_type(8)))  unsigned int u32x8;

// Build a 16-element bf16 fragment from two contiguous 8-element (16B) chunks.
__device__ __forceinline__ v16bf frag16(const bf16_t* p0, const bf16_t* p1) {
  v8bf lo = *(const v8bf*)p0;
  v8bf hi = *(const v8bf*)p1;
  return __builtin_shufflevector(lo, hi, 0,1,2,3,4,5,6,7,8,9,10,11,12,13,14,15);
}

__device__ __forceinline__ v8f wmma_bf16(v16bf a, v16bf b, v8f c) {
  return __builtin_amdgcn_wmma_f32_16x16x32_bf16(false, a, false, b, (short)0, c,
                                                 false, false);
}

__device__ __forceinline__ unsigned rfl(unsigned v) {
  return (unsigned)__builtin_amdgcn_readfirstlane((int)v);
}

// Issue one TDM 2D tile load: tile_dim0=32 elems (64B rows), tile_dim1=64 rows,
// 2-byte elements, LDS row padding 16B -> row stride 80B (40 bf16).
// OOB rows (beyond tdim1) are zero-filled by the TDM per the D# spec.
__device__ __forceinline__ void tdm_load_tile_64x32(
    unsigned lds_byte_addr, unsigned long long gaddr,
    unsigned tdim0, unsigned tdim1, unsigned stride0_elems) {
  u32x4 g0;
  g0[0] = rfl(1u);                                    // count=1, user mode
  g0[1] = rfl(lds_byte_addr);                         // lds_addr
  g0[2] = rfl((unsigned)gaddr);                       // global_addr[31:0]
  g0[3] = rfl(((unsigned)(gaddr >> 32) & 0x1FFFFFFu)  // global_addr[56:32]
              | (2u << 30));                          // type = 2 ("image")
  u32x8 g1;
  g1[0] = rfl((1u << 16)      // data_size = 1 (2 bytes)
              | (1u << 20)    // pad_enable
              | (3u << 22)    // pad_interval: 16 DWORDs between pads
              | (3u << 25));  // pad_amount: 4 DWORDs (16B) pad
  g1[1] = rfl((tdim0 & 0xFFFFu) << 16);               // tensor_dim0[15:0]
  g1[2] = rfl((tdim0 >> 16) | ((tdim1 & 0xFFFFu) << 16)); // td0[31:16]|td1[15:0]
  g1[3] = rfl((tdim1 >> 16) | (32u << 16));           // td1[31:16] | tile_dim0=32
  g1[4] = rfl(64u);                                   // tile_dim1=64, tile_dim2=0
  g1[5] = rfl(stride0_elems);                         // tensor_dim0_stride[31:0]
  g1[6] = rfl(0u);                                    // stride0[47:32] | stride1
  g1[7] = rfl(0u);
  asm volatile("tensor_load_to_lds %0, %1" :: "s"(g0), "s"(g1) : "memory");
}

// ---------------------------------------------------------------------------
// Implicit-GEMM conv + fused BN(scale,bias) + residual + LReLU epilogue.
// GEMM view: M = Cout, N = out pixels (per image), K = Cin*KH*KW (padded to 32).
// Workgroup = 8 waves -> 64x64 output tile; each wave owns one 16-row A strip
// and two 16x16 WMMA tiles (A fragment reused). LDS double buffered; the A tile
// (packed weights) is fetched by the Tensor Data Mover, overlapped with WMMA.
// ---------------------------------------------------------------------------
template <int KHKW>
__global__ __launch_bounds__(256) void conv_wmma_t(
    const bf16_t* __restrict__ in, const bf16_t* __restrict__ wpk,
    const float* __restrict__ scale, const float* __restrict__ bias,
    const bf16_t* __restrict__ resid, bf16_t* __restrict__ out,
    float* __restrict__ outf32, int f32_nlimit,
    int Cin, int H, int W, int Cout,
    int stride, int pad, int Ho, int woLog, int Kpad, int do_lrelu)
{
  constexpr int KW = (KHKW == 9) ? 3 : 1;
  __shared__ bf16_t Asub[2][64 * 40];  // [64 cout][32 k], row stride 40 (TDM pad)
  __shared__ bf16_t Bsub[2][64 * 40];  // [64 pix ][32 k], row stride 40

  const int n        = blockIdx.z;
  const int pixBase  = blockIdx.x * 64;
  const int coutBase = blockIdx.y * 64;
  const int tid      = threadIdx.x;
  const int lane     = tid & 31;
  const int wv       = tid >> 5;
  const int mt       = wv >> 1;          // cout sub-tile 0..3
  const int ntB      = (wv & 1) * 2;     // first of two pixel sub-tiles
  const int lanelow  = lane & 15;
  const int half     = lane >> 4;
  const int Wo       = 1 << woLog;
  const int HoWo     = Ho << woLog;
  const int HW       = H * W;
  const bool fast1x1 = (KHKW == 1) && (stride == 1) && (pad == 0);
  // LDS byte offset of Asub (generic LDS pointers carry the offset in the
  // low 32 bits; the shared aperture base has zero low bits).
  const unsigned ldsA0 = (unsigned)(size_t)(&Asub[0][0]);

  auto stage = [&](int buf, int kc) {
    // ---- A: packed weights via Tensor Data Mover (one issue per chunk) ----
    if (wv == 0) {
      tdm_load_tile_64x32(
          ldsA0 + (unsigned)buf * (64 * 40 * 2),
          (unsigned long long)(size_t)(wpk + (size_t)coutBase * Kpad + kc),
          (unsigned)(Kpad - kc), (unsigned)(Cout - coutBase), (unsigned)Kpad);
    }
    // ---- B: im2col^T [64 pix][32 k] ----
    if (fast1x1) {
      // B^T[p][k] = in[n][kc+k][pixBase+p]: pixel-contiguous -> 16B loads.
      int kk  = tid & 31;
      int pg8 = (tid >> 5) * 8;                  // 8 pixels per thread
      int ci  = kc + kk;
      uint4 v; v.x = 0u; v.y = 0u; v.z = 0u; v.w = 0u;
      if (ci < Cin && pixBase + pg8 < HoWo)
        v = *(const uint4*)(in + ((size_t)n * Cin + ci) * HW + pixBase + pg8);
      const bf16_t* pv = (const bf16_t*)&v;
#pragma unroll
      for (int q = 0; q < 8; ++q)
        Bsub[buf][(pg8 + q) * 40 + kk] = pv[q];
    } else {
#pragma unroll
      for (int it = 0; it < 8; ++it) {
        int e  = tid + it * 256;                 // 2048 elements
        int p  = e >> 5;
        int kk = e & 31;
        int k  = kc + kk;
        int ci = k / KHKW;                       // compile-time divisor
        int rr = k - ci * KHKW;
        int kh = rr / KW;
        int kw = rr - kh * KW;
        bf16_t v = (bf16_t)0.0f;
        int pg = pixBase + p;
        if (pg < HoWo && ci < Cin) {
          int oh = pg >> woLog;
          int ow = pg & (Wo - 1);
          int ih = oh * stride - pad + kh;
          int iw = ow * stride - pad + kw;
          if (ih >= 0 && ih < H && iw >= 0 && iw < W)
            v = in[((size_t)n * Cin + ci) * HW + ih * W + iw];
        }
        Bsub[buf][p * 40 + kk] = v;
      }
    }
  };

  v8f acc0 = {0.f,0.f,0.f,0.f,0.f,0.f,0.f,0.f};
  v8f acc1 = {0.f,0.f,0.f,0.f,0.f,0.f,0.f,0.f};

  stage(0, 0);
  __builtin_amdgcn_s_wait_tensorcnt(0);
  __syncthreads();
  int buf = 0;
  for (int kc = 0; kc < Kpad; kc += 32) {
    if (kc + 32 < Kpad) stage(buf ^ 1, kc + 32);   // overlap with WMMA below
    const bf16_t* arow  = &Asub[buf][(mt * 16 + lanelow) * 40];
    const bf16_t* brow0 = &Bsub[buf][(ntB * 16 + lanelow) * 40];
    const bf16_t* brow1 = brow0 + 16 * 40;
    v16bf a  = frag16(arow + half * 8,   arow + 16 + half * 8);
    v16bf b0 = frag16(brow0 + half * 8, brow0 + 16 + half * 8);
    v16bf b1 = frag16(brow1 + half * 8, brow1 + 16 + half * 8);
    acc0 = wmma_bf16(a, b0, acc0);
    acc1 = wmma_bf16(a, b1, acc1);
    __builtin_amdgcn_s_wait_tensorcnt(0);
    buf ^= 1;
    __syncthreads();
  }

  // ---- epilogue: BN + residual + LReLU, bf16 store (+ optional f32 copy) ----
  const int p0 = pixBase + ntB * 16 + lanelow;
  const int p1 = p0 + 16;
#pragma unroll
  for (int r = 0; r < 8; ++r) {
    int co = coutBase + mt * 16 + r + 8 * half;
    if (co < Cout) {
      float sc = scale[co], bi = bias[co];
      size_t rowo = ((size_t)n * Cout + co) * HoWo;
      if (p0 < HoWo) {
        float v = acc0[r] * sc + bi;
        if (resid) v += (float)resid[rowo + p0];
        if (do_lrelu) v = v > 0.f ? v : 0.2f * v;
        out[rowo + p0] = (bf16_t)v;
        if (outf32 && n < f32_nlimit) outf32[rowo + p0] = v;
      }
      if (p1 < HoWo) {
        float v = acc1[r] * sc + bi;
        if (resid) v += (float)resid[rowo + p1];
        if (do_lrelu) v = v > 0.f ? v : 0.2f * v;
        out[rowo + p1] = (bf16_t)v;
        if (outf32 && n < f32_nlimit) outf32[rowo + p1] = v;
      }
    }
  }
}

// ---------------------------------------------------------------------------
// cov_j = Xc_j * Xc_j^T / 511 ; Xc [5][256 c][512 n] bf16, K contiguous.
// Direct-from-global fragments (rows are 512 bf16 contiguous).
// ---------------------------------------------------------------------------
__global__ __launch_bounds__(256) void cov_wmma(const bf16_t* __restrict__ Xc,
                                                bf16_t* __restrict__ cov)
{
  const int j    = blockIdx.y;
  const int tid  = threadIdx.x;
  const int lane = tid & 31, wv = tid >> 5;
  const int t    = blockIdx.x * 8 + wv;       // 256 tiles per class
  const int mt   = t >> 4, nt = t & 15;
  const int lanelow = lane & 15, half = lane >> 4;

  const bf16_t* Xj   = Xc + (size_t)j * 256 * 512;
  const bf16_t* arow = Xj + (size_t)(mt * 16 + lanelow) * 512;
  const bf16_t* brow = Xj + (size_t)(nt * 16 + lanelow) * 512;

  v8f acc = {0.f, 0.f, 0.f, 0.f, 0.f, 0.f, 0.f, 0.f};
#pragma unroll 4
  for (int kc = 0; kc < 512; kc += 32) {
    v16bf a = frag16(arow + kc + half * 8,  arow + kc + 16 + half * 8);
    v16bf b = frag16(brow + kc + half * 8,  brow + kc + 16 + half * 8);
    acc = wmma_bf16(a, b, acc);
  }
  const float s = 1.0f / 511.0f;
#pragma unroll
  for (int r = 0; r < 8; ++r) {
    int m = mt * 16 + r + 8 * half;
    cov[((size_t)j * 256 + m) * 256 + nt * 16 + lanelow] = (bf16_t)(acc[r] * s);
  }
}

// ---------------------------------------------------------------------------
// sim[b,j,p] = sum_{c,d} q[b,c,p] cov[j,c,d] q[b,d,p]; then LReLU and dot
// with cova_w over p -> cova_score[b,j]. One workgroup per (b,j).
// ---------------------------------------------------------------------------
__global__ __launch_bounds__(256) void sim_cova(
    const bf16_t* __restrict__ cov, const bf16_t* __restrict__ qT,
    const float* __restrict__ cova_w, const float* __restrict__ cova_b,
    float* __restrict__ cova_out)
{
  const int b = blockIdx.x, j = blockIdx.y;
  __shared__ float simLDS[64];
  __shared__ float red[64];
  const int tid = threadIdx.x;
  if (tid < 64) simLDS[tid] = 0.f;
  __syncthreads();

  const int lane = tid & 31, wv = tid >> 5;
  const int lanelow = lane & 15, half = lane >> 4;
  const int nt = wv & 3;            // fixed pixel sub-tile per wave
  const int mb = wv >> 2;           // mt = mb + 2*i
  const bf16_t* covj = cov + (size_t)j * 256 * 256;
  const bf16_t* qtb  = qT + (size_t)b * 64 * 256;
  const int p = nt * 16 + lanelow;
  const bf16_t* brow = qtb + (size_t)p * 256;

  float psum = 0.f;
  for (int i = 0; i < 8; ++i) {
    int mt = mb + 2 * i;
    const bf16_t* arow = covj + (size_t)(mt * 16 + lanelow) * 256;
    v8f acc = {0.f, 0.f, 0.f, 0.f, 0.f, 0.f, 0.f, 0.f};
#pragma unroll
    for (int kc = 0; kc < 256; kc += 32) {
      v16bf a  = frag16(arow + kc + half * 8,  arow + kc + 16 + half * 8);
      v16bf b2 = frag16(brow + kc + half * 8,  brow + kc + 16 + half * 8);
      acc = wmma_bf16(a, b2, acc);
    }
#pragma unroll
    for (int r = 0; r < 8; ++r) {
      int d = mt * 16 + r + 8 * half;
      psum += acc[r] * (float)qtb[(size_t)p * 256 + d];
    }
  }
  atomicAdd(&simLDS[p], psum);
  __syncthreads();

  if (tid < 64) {
    float s = simLDS[tid];
    s = s > 0.f ? s : 0.2f * s;                // LReLU before Conv1d fold
    red[tid] = s * cova_w[tid];
  }
  __syncthreads();
  if (tid == 0) {
    float a = cova_b[0];
    for (int i = 0; i < 64; ++i) a += red[i];
    cova_out[b * 5 + j] = a;
  }
}

// ---------------------------------------------------------------------------
// Small utility kernels
// ---------------------------------------------------------------------------
__global__ void f32_to_bf16_k(const float* __restrict__ src,
                              bf16_t* __restrict__ dst, int n) {
  int i = blockIdx.x * 256 + threadIdx.x;
  if (i < n) dst[i] = (bf16_t)src[i];
}

__global__ void pack_weights_k(const float* __restrict__ w,
                               bf16_t* __restrict__ dst,
                               int Cout, int K, int Kpad) {
  int i = blockIdx.x * 256 + threadIdx.x;
  if (i < Cout * Kpad) {
    int co = i / Kpad, k = i - co * Kpad;
    dst[i] = (k < K) ? (bf16_t)w[co * K + k] : (bf16_t)0.0f;
  }
}

__global__ void bn_prep_k(const float* __restrict__ g, const float* __restrict__ b,
                          const float* __restrict__ m, const float* __restrict__ v,
                          float* __restrict__ scale, float* __restrict__ bias, int C) {
  int i = blockIdx.x * 64 + threadIdx.x;
  if (i < C) {
    float s = g[i] * rsqrtf(v[i] + 1e-5f);
    scale[i] = s;
    bias[i]  = b[i] - m[i] * s;
  }
}

// gfeat = mean over 64 pixels; cls = gfeat @ fc_w^T + fc_b
__global__ __launch_bounds__(256) void pool_fc_k(const float* __restrict__ feat,
                                                 const float* __restrict__ fcw,
                                                 const float* __restrict__ fcb,
                                                 float* __restrict__ cls) {
  __shared__ float g[256];
  const int b = blockIdx.x, c = threadIdx.x;
  const float* f = feat + ((size_t)b * 256 + c) * 64;
  float s = 0.f;
  for (int p = 0; p < 64; ++p) s += f[p];
  g[c] = s * (1.0f / 64.0f);
  __syncthreads();
  if (c < 10) {
    float a = fcb[c];
    const float* wr = fcw + c * 256;
    for (int k = 0; k < 256; ++k) a += g[k] * wr[k];
    cls[b * 10 + c] = a;
  }
}

// per-(b,c) L2-normalize q over the 64 pixels and store transposed [b][p][c] bf16
__global__ __launch_bounds__(256) void qnorm_transpose_k(const float* __restrict__ feat,
                                                         bf16_t* __restrict__ qT) {
  const int b = blockIdx.x, c = threadIdx.x;
  const float* f = feat + ((size_t)b * 256 + c) * 64;
  float s = 0.f;
  for (int p = 0; p < 64; ++p) s += f[p] * f[p];
  float rn = rsqrtf(fmaxf(s, 1e-30f));
  for (int p = 0; p < 64; ++p)
    qT[((size_t)b * 64 + p) * 256 + c] = (bf16_t)(f[p] * rn);
}

// X[j][c][n] = feat_s; Xc = X - mean_n(X). featbf: [296][256][64] bf16, support at n>=256
__global__ __launch_bounds__(128) void build_xc_k(const bf16_t* __restrict__ featbf,
                                                  bf16_t* __restrict__ Xc) {
  const int j = blockIdx.x >> 8;
  const int c = blockIdx.x & 255;
  const bf16_t* base = featbf + ((size_t)(256 + j * 8) * 256 + c) * 64;
  __shared__ float part[128];
  const int tid = threadIdx.x;
  float s = 0.f;
  for (int n = tid; n < 512; n += 128) {
    int sh = n >> 6, p = n & 63;
    s += (float)base[(size_t)sh * 256 * 64 + p];
  }
  part[tid] = s;
  __syncthreads();
  for (int off = 64; off > 0; off >>= 1) {
    if (tid < off) part[tid] += part[tid + off];
    __syncthreads();
  }
  float mean = part[0] * (1.0f / 512.0f);
  for (int n = tid; n < 512; n += 128) {
    int sh = n >> 6, p = n & 63;
    Xc[((size_t)j * 256 + c) * 512 + n] =
        (bf16_t)((float)base[(size_t)sh * 256 * 64 + p] - mean);
  }
}

// ---------------------------------------------------------------------------
// Host orchestration
// ---------------------------------------------------------------------------
extern "C" void kernel_launch(void* const* d_in, const int* in_sizes, int n_in,
                              void* d_out, int out_size, void* d_ws, size_t ws_size,
                              hipStream_t stream) {
  (void)in_sizes; (void)n_in; (void)out_size; (void)ws_size;
  const float* x   = (const float*)d_in[0];
  const float* sup = (const float*)d_in[1];

  // --- params flattened in sorted-key pytree order ---
  int pi = 2;
  auto F = [&](int i) { return (const float*)d_in[i]; };
  const float* cova_b = F(pi++);
  const float* cova_w = F(pi++);
  const float* fc_b   = F(pi++);
  const float* fc_w   = F(pi++);
  struct BNP { const float *b, *g, *m, *v; };
  auto getBN = [&]() { BNP r{F(pi), F(pi + 1), F(pi + 2), F(pi + 3)}; pi += 4; return r; };
  BNP bn1[3][6], bn2[3][6], bn3[3][6], bnd[3];
  const float *w1s[3][6], *w2s[3][6], *w3s[3][6], *wds[3];
  for (int L = 0; L < 3; ++L)
    for (int i = 0; i < 6; ++i) {
      bn1[L][i] = getBN(); bn2[L][i] = getBN(); bn3[L][i] = getBN();
      if (i == 0) bnd[L] = getBN();
      w1s[L][i] = F(pi++); w2s[L][i] = F(pi++); w3s[L][i] = F(pi++);
      if (i == 0) wds[L] = F(pi++);
    }
  BNP stem_bn = getBN();
  const float* stem_w = F(pi++);

  // --- workspace carve (~160 MB total) ---
  char* base = (char*)d_ws;
  size_t off = 0;
  auto carve = [&](size_t bytes) -> char* {
    char* r = base + off;
    off = (off + bytes + 255) & ~(size_t)255;
    return r;
  };
  bf16_t* wpkA = (bf16_t*)carve((size_t)4 << 20);
  float*  bnA  = (float*) carve((size_t)256 << 10);
  bf16_t* xin  = (bf16_t*)carve((size_t)296 * 3 * 1024 * 2);
  bf16_t* bufA = (bf16_t*)carve((size_t)296 * 64 * 1024 * 2);
  bf16_t* bufB = (bf16_t*)carve((size_t)296 * 64 * 1024 * 2);
  bf16_t* T1   = (bf16_t*)carve((size_t)296 * 32 * 1024 * 2);
  bf16_t* T2   = (bf16_t*)carve((size_t)296 * 16 * 1024 * 2);
  bf16_t* IDb  = (bf16_t*)carve((size_t)296 * 64 * 1024 * 2);
  bf16_t* XcB  = (bf16_t*)carve((size_t)5 * 256 * 512 * 2);
  bf16_t* covB = (bf16_t*)carve((size_t)5 * 256 * 256 * 2);
  bf16_t* qTB  = (bf16_t*)carve((size_t)256 * 64 * 256 * 2);

  // --- weight packing + BN folding ---
  size_t wElems = 0, bnElems = 0;
  struct CW { const bf16_t* w; const float* sc; const float* bi; int Kpad; int khkw; };
  auto packConv = [&](const float* wsrc, int Cout, int K, int khkw, BNP bn) -> CW {
    int Kpad = (K + 31) & ~31;
    bf16_t* dst = wpkA + wElems;
    wElems += (size_t)Cout * Kpad;
    int n = Cout * Kpad;
    pack_weights_k<<<(n + 255) / 256, 256, 0, stream>>>(wsrc, dst, Cout, K, Kpad);
    float* sc = bnA + bnElems; bnElems += Cout;
    float* bi = bnA + bnElems; bnElems += Cout;
    bn_prep_k<<<(Cout + 63) / 64, 64, 0, stream>>>(bn.g, bn.b, bn.m, bn.v, sc, bi, Cout);
    return {dst, sc, bi, Kpad, khkw};
  };

  const int NIMG = 296;
  float* O = (float*)d_out;
  float* clsOut  = O;
  float* covaOut = O + 2560;
  float* featF   = O + 3840;

  int planes[3]  = {16, 32, 64};
  int cin0[3]    = {16, 64, 128};
  int strideL[3] = {1, 2, 2};

  CW stemC = packConv(stem_w, 16, 27, 9, stem_bn);
  CW c1[3][6], c2[3][6], c3[3][6], cd[3];
  for (int L = 0; L < 3; ++L)
    for (int i = 0; i < 6; ++i) {
      int cin = (i == 0) ? cin0[L] : planes[L] * 4;
      c1[L][i] = packConv(w1s[L][i], planes[L], cin, 1, bn1[L][i]);
      c2[L][i] = packConv(w2s[L][i], planes[L], planes[L] * 9, 9, bn2[L][i]);
      c3[L][i] = packConv(w3s[L][i], planes[L] * 4, planes[L], 1, bn3[L][i]);
      if (i == 0) cd[L] = packConv(wds[L], planes[L] * 4, cin, 1, bnd[L]);
    }

  // --- convert inputs (queries then support -> combined batch of 296) ---
  { int n = 256 * 3 * 1024;
    f32_to_bf16_k<<<(n + 255) / 256, 256, 0, stream>>>(x, xin, n); }
  { int n = 40 * 3 * 1024;
    f32_to_bf16_k<<<(n + 255) / 256, 256, 0, stream>>>(sup, xin + (size_t)256 * 3 * 1024, n); }

  auto ilog2 = [](int v) { int l = 0; while ((1 << (l + 1)) <= v) ++l; return l; };

  auto conv = [&](const bf16_t* in, int Cin, int H, CW cw, int Cout, int st,
                  int pd, const bf16_t* resid, bf16_t* outp, float* outf, int f32lim,
                  int act) {
    int KH = (cw.khkw == 9) ? 3 : 1;
    int Ho = (H + 2 * pd - KH) / st + 1;
    int woLog = ilog2(Ho);
    dim3 grid((Ho * Ho + 63) / 64, (Cout + 63) / 64, NIMG);
    if (cw.khkw == 1)
      conv_wmma_t<1><<<grid, 256, 0, stream>>>(
          in, cw.w, cw.sc, cw.bi, resid, outp, outf, f32lim,
          Cin, H, H, Cout, st, pd, Ho, woLog, cw.Kpad, act);
    else
      conv_wmma_t<9><<<grid, 256, 0, stream>>>(
          in, cw.w, cw.sc, cw.bi, resid, outp, outf, f32lim,
          Cin, H, H, Cout, st, pd, Ho, woLog, cw.Kpad, act);
  };

  // --- backbone ---
  conv(xin, 3, 32, stemC, 16, 1, 1, nullptr, bufA, nullptr, 0, 1);
  bf16_t* cur = bufA;
  bf16_t* nxt = bufB;
  int H = 32;
  for (int L = 0; L < 3; ++L) {
    for (int i = 0; i < 6; ++i) {
      int st  = (i == 0) ? strideL[L] : 1;
      int cin = (i == 0) ? cin0[L] : planes[L] * 4;
      int p4  = planes[L] * 4;
      int Ho  = H / st;
      const bf16_t* resid;
      if (i == 0) {
        conv(cur, cin, H, cd[L], p4, st, 0, nullptr, IDb, nullptr, 0, 0);
        resid = IDb;
      } else {
        resid = cur;
      }
      conv(cur, cin, H, c1[L][i], planes[L], 1, 0, nullptr, T1, nullptr, 0, 1);
      conv(T1, planes[L], H, c2[L][i], planes[L], st, 1, nullptr, T2, nullptr, 0, 1);
      bool last = (L == 2 && i == 5);
      conv(T2, planes[L], Ho, c3[L][i], p4, 1, 0, resid, nxt,
           last ? featF : nullptr, 256, 1);
      bf16_t* t = cur; cur = nxt; nxt = t;
      H = Ho;
    }
  }

  // --- heads ---
  pool_fc_k<<<256, 256, 0, stream>>>(featF, fc_w, fc_b, clsOut);
  qnorm_transpose_k<<<256, 256, 0, stream>>>(featF, qTB);
  build_xc_k<<<1280, 128, 0, stream>>>(cur, XcB);
  cov_wmma<<<dim3(32, 5), 256, 0, stream>>>(XcB, covB);
  sim_cova<<<dim3(256, 5), 256, 0, stream>>>(covB, qTB, cova_w, cova_b, covaOut);
}
